// AutoCorrelation_28338194219699
// MI455X (gfx1250) — matrix operand order repaired and checked by reading, not compiled, not run on hardware
//
#include <hip/hip_runtime.h>
#include <math.h>

typedef __attribute__((ext_vector_type(16))) _Float16 v16h;
typedef __attribute__((ext_vector_type(8)))  _Float16 v8h;
typedef __attribute__((ext_vector_type(8)))  float    v8f;

#define LEN   4096
#define LMASK 4095
#define TPB   256
#define K8    8

// Load 8 consecutive (circular) f16 row elements starting at index a as one
// 16-byte-aligned LDS b128 load, using the 8 shift-replicated copies.
__device__ __forceinline__ v8h lds_load8(const _Float16* xrep, int a) {
  a &= LMASK;                  // circular wrap (two's complement & = mod 4096)
  const int s = a & 7;         // which shifted copy
  const int q = a & ~7;        // 8-element-aligned offset inside the copy
  return *(const v8h*)(xrep + (s << 12) + q);   // s*LEN + q, 16B aligned
}

#define CONCAT16(lo, hi) __builtin_shufflevector((lo), (hi), \
    0,1,2,3,4,5,6,7,8,9,10,11,12,13,14,15)

__launch_bounds__(TPB, 1)
__global__ void autocorr_fused_kernel(const float* __restrict__ x,
                                      float* __restrict__ out) {
  __shared__ __align__(16) unsigned char smem[96 * 1024 + 64];
  _Float16* xrep = (_Float16*)smem;                  // 8 shifted f16 copies, 64KB
  float*    xf   = (float*)(smem + 64 * 1024);       // f32 row, 16KB
  float*    corr = (float*)(smem + 80 * 1024);       // autocorr result, 16KB
  float*    s_w  = (float*)(smem + 96 * 1024);       // 8 softmax weights
  int*      s_d  = (int*)  (smem + 96 * 1024 + 32);  // 8 delays
  // top-k scratch aliases xrep (dead after phase 1, separated by barrier)
  float (*cvals)[K8] = (float(*)[K8])(smem);
  int   (*cidx )[K8] = (int  (*)[K8])(smem + 8 * 1024);

  const int tid  = threadIdx.x;
  const int lane = tid & 31;
  const int wave = tid >> 5;
  const int jm   = lane & 15;        // N col for B / M row i for A / D col
  const int hi   = lane >> 4;        // which half of the wave
  const size_t rowbase = (size_t)blockIdx.x * LEN;

  // ---------- stage row into LDS: f32 copy + 8 shifted f16 copies ----------
  for (int i = tid; i < LEN; i += TPB) {
    float v = x[rowbase + i];
    xf[i] = v;
    _Float16 h = (_Float16)v;
    #pragma unroll
    for (int s = 0; s < 8; ++s)            // xrep[s][(i-s)&M] = x[i]
      xrep[(s << 12) + ((i - s) & LMASK)] = h;
  }
  __syncthreads();

  // ---------- phase 1: circular autocorrelation via WMMA ----------
  // corr[256m + 16i + j] = sum_u x[u - 16i - 256m] * x[u + j]
  const int m0 = wave * 2;
  const int m1 = m0 + 1;
  const int bbase_l = 8 * hi + jm;                    // + u0
  const int abase_l = -16 * jm - (m0 << 8) + 8 * hi;  // + u0
  v8f acc0 = {};
  v8f acc1 = {};
  #pragma unroll 2
  for (int u0 = 0; u0 < LEN; u0 += 32) {
    const int bb = u0 + bbase_l;
    v16h bfrag = CONCAT16(lds_load8(xrep, bb), lds_load8(xrep, bb + 16));

    const int a0 = u0 + abase_l;            // tile m0
    v16h afrag0 = CONCAT16(lds_load8(xrep, a0), lds_load8(xrep, a0 + 16));
    const int a1 = a0 - 256;                // tile m1
    v16h afrag1 = CONCAT16(lds_load8(xrep, a1), lds_load8(xrep, a1 + 16));

    acc0 = __builtin_amdgcn_wmma_f32_16x16x32_f16(false, afrag0, false, bfrag,
                                                  (short)0, acc0, false, false);
    acc1 = __builtin_amdgcn_wmma_f32_16x16x32_f16(false, afrag1, false, bfrag,
                                                  (short)0, acc1, false, false);
  }
  // D layout: VGPR r -> M = r + 8*hi, N = jm ; tau = 256m + 16*M + N
  #pragma unroll
  for (int r = 0; r < 8; ++r) {
    corr[(m0 << 8) + ((r + 8 * hi) << 4) + jm] = acc0[r];
    corr[(m1 << 8) + ((r + 8 * hi) << 4) + jm] = acc1[r];
  }
  __syncthreads();   // corr complete; xrep now dead -> cvals/cidx may reuse it

  // ---------- phase 2: top-8 of corr ----------
  float bv[K8]; int bix[K8];
  #pragma unroll
  for (int k = 0; k < K8; ++k) { bv[k] = -3.402823e38f; bix[k] = 0x7fffffff; }
  for (int i = tid; i < LEN; i += TPB) {
    float v  = corr[i];
    int   vi = i;
    #pragma unroll
    for (int k = 0; k < K8; ++k) {   // sorted (desc) bubble insertion
      bool gt = (v > bv[k]) || (v == bv[k] && vi < bix[k]);
      float tv = bv[k]; int ti = bix[k];
      if (gt) { bv[k] = v; bix[k] = vi; v = tv; vi = ti; }
    }
  }
  #pragma unroll
  for (int k = 0; k < K8; ++k) { cvals[tid][k] = bv[k]; cidx[tid][k] = bix[k]; }
  __syncthreads();

  // tree merge: Batcher elementwise-max + bitonic clean (all static indexing)
  for (int s = TPB >> 1; s >= 1; s >>= 1) {
    if (tid < s) {
      float v[K8]; int ix[K8];
      #pragma unroll
      for (int k = 0; k < K8; ++k) {
        float a = cvals[tid][k];         int ai = cidx[tid][k];
        float b = cvals[tid + s][7 - k]; int bi = cidx[tid + s][7 - k];
        bool ta = (a > b) || (a == b && ai < bi);
        v[k] = ta ? a : b; ix[k] = ta ? ai : bi;
      }
      #define CE(p,q) { bool sw_ = (v[p] < v[q]) || (v[p] == v[q] && ix[p] > ix[q]); \
                        if (sw_) { float tv_ = v[p]; v[p] = v[q]; v[q] = tv_;        \
                                   int ti_ = ix[p]; ix[p] = ix[q]; ix[q] = ti_; } }
      CE(0,4) CE(1,5) CE(2,6) CE(3,7)
      CE(0,2) CE(1,3) CE(4,6) CE(5,7)
      CE(0,1) CE(2,3) CE(4,5) CE(6,7)
      #undef CE
      #pragma unroll
      for (int k = 0; k < K8; ++k) { cvals[tid][k] = v[k]; cidx[tid][k] = ix[k]; }
    }
    __syncthreads();
  }

  // ---------- softmax over top-8 ----------
  if (tid == 0) {
    float mx = cvals[0][0];            // sorted desc -> element 0 is max
    float sum = 0.f; float w[K8];
    #pragma unroll
    for (int k = 0; k < K8; ++k) { w[k] = __expf(cvals[0][k] - mx); sum += w[k]; }
    float inv = 1.f / sum;
    #pragma unroll
    for (int k = 0; k < K8; ++k) { s_w[k] = w[k] * inv; s_d[k] = cidx[0][k]; }
  }
  __syncthreads();

  // ---------- phase 3: delay aggregation ----------
  float w[K8]; int d[K8];
  #pragma unroll
  for (int k = 0; k < K8; ++k) { w[k] = s_w[k]; d[k] = s_d[k]; }
  for (int t = tid; t < LEN; t += TPB) {
    float acc = 0.f;
    #pragma unroll
    for (int k = 0; k < K8; ++k)
      acc += w[k] * xf[(t + d[k]) & LMASK];
    out[rowbase + t] = acc;
  }
}

extern "C" void kernel_launch(void* const* d_in, const int* in_sizes, int n_in,
                              void* d_out, int out_size, void* d_ws, size_t ws_size,
                              hipStream_t stream) {
  (void)n_in; (void)d_ws; (void)ws_size; (void)out_size;
  const float* x = (const float*)d_in[0];
  float* out = (float*)d_out;
  const int rows = in_sizes[0] / LEN;   // 8 * 512 = 4096 rows
  autocorr_fused_kernel<<<rows, TPB, 0, stream>>>(x, out);
}
